// FrameTransformerDecoder_26680336843187
// MI455X (gfx1250) — matrix operand order repaired and checked by reading, not compiled, not run on hardware
//
#include <hip/hip_runtime.h>
#include <cstdint>

// ---------------------------------------------------------------------------
// CDNA5 / gfx1250 implementation of the FrameTransformerDecoder block.
// All heavy math runs on v_wmma_f32_16x16x32_bf16. The GEMM is double-buffered:
// A tiles stream through the CDNA5 async copy engine (global_load_async_to_lds
// _b128 / ASYNCcnt) and B tiles through VGPRs (packed to single ds_store_b128
// per octet), both for tile k+1 while tile k feeds the WMMA pipe.
// ---------------------------------------------------------------------------

typedef __attribute__((ext_vector_type(16))) __bf16        v16bf;
typedef __attribute__((ext_vector_type(8)))  float         v8f;
typedef __attribute__((ext_vector_type(8)))  unsigned int  v8u;
typedef unsigned short u16;

struct alignas(16) U4 { unsigned int x, y, z, w; };

#define CDIM 8
#define FEATD 512
#define WDIM 1024
#define NHD  8
#define HD   64
#define GEXP 2048

__device__ __forceinline__ u16 f2bf(float f) {
  unsigned int u = __builtin_bit_cast(unsigned int, f);
  u += 0x7FFFu + ((u >> 16) & 1u);          // round-to-nearest-even
  return (u16)(u >> 16);
}

// ---------------------------------------------------------------------------
// Batched WMMA GEMM:  C[z] = alpha * A[z] @ B[z] (+ Res)
//   A: M x K bf16 row-major
//   B: K x N bf16 row-major  (or N x K row-major when B_NK, e.g. K^T for QK)
//   C: M x N f32
// Block: 256 threads (8 waves). Tile: 128 x (NT*16). K-step: 32.
// Double-buffered: async A copy + B global loads for step s+1 are issued
// before the WMMA chain of step s; s_wait_asynccnt + barrier come after it.
// Rows padded to 80 B: 16 B alignment for b128 ops, 20-bank stride => no
// LDS bank conflicts on fragment reads. B staging: each thread owns
// (n, k-octet) pairs so the transposed write is one ds_store_b128.
// ---------------------------------------------------------------------------
template <bool B_NK, int NT>
__global__ void __launch_bounds__(256)
gemm_wmma(const u16* __restrict__ Ap, const u16* __restrict__ Bp,
          float* __restrict__ Cp, const float* __restrict__ Rp,
          int M, int N, int K, long sA, long sB, long sC, float alpha) {
  constexpr int BN   = NT * 16;
  constexpr int NOCT = BN / 64;            // (32*BN)/(256 threads * 8 elems)

  __shared__ __align__(16) u16 As[2][128][40];
  __shared__ __align__(16) u16 Bs[2][BN][40];

  const int t  = threadIdx.x;
  const int z  = blockIdx.z;
  const int m0 = blockIdx.y * 128;
  const int n0 = blockIdx.x * BN;
  const int wv = t >> 5;        // wave id (0..7) -> M sub-tile
  const int l  = t & 31;        // lane
  const int hf = l >> 4;        // lane half
  const int lr = l & 15;

  // ---- staging helpers -----------------------------------------------------
  auto asyncA = [&](int kk, int buf) {   // 128x32 bf16, 2 x 16B chunks/thread
#pragma unroll
    for (int i = 0; i < 2; i++) {
      int idx = i * 256 + t;            // 0..511
      int r = idx >> 2, ch = idx & 3;   // row, 16B chunk in row
      const u16* gp = Ap + (long)z * sA + (long)(m0 + r) * K + kk + ch * 8;
      unsigned int lds = (unsigned int)(unsigned long long)&As[buf][r][ch * 8];
      asm volatile("global_load_async_to_lds_b128 %0, %1, off"
                   :: "v"(lds), "v"((unsigned long long)gp) : "memory");
    }
  };
  // Each thread owns NOCT (n, k-octet) pairs: 8 consecutive k for a fixed n.
  auto loadB = [&](int kk, u16 (*regs)[8]) {
#pragma unroll
    for (int i = 0; i < NOCT; i++) {
      int idx = i * 256 + t;
      int n = idx % BN, ko = (idx / BN) * 8;
      if (B_NK) {
        // contiguous 16 B in global: compiler merges into a wide load
#pragma unroll
        for (int j = 0; j < 8; j++)
          regs[i][j] = Bp[(long)z * sB + (long)(n0 + n) * K + kk + ko + j];
      } else {
        // fixed j => lanes sweep consecutive n: coalesced wave access
#pragma unroll
        for (int j = 0; j < 8; j++)
          regs[i][j] = Bp[(long)z * sB + (long)(kk + ko + j) * N + n0 + n];
      }
    }
  };
  auto storeB = [&](int buf, u16 (*regs)[8]) {   // transposed: Bs[n][k]
#pragma unroll
    for (int i = 0; i < NOCT; i++) {
      int idx = i * 256 + t;
      int n = idx % BN, ko = (idx / BN) * 8;
      U4 p;
      p.x = (unsigned)regs[i][0] | ((unsigned)regs[i][1] << 16);
      p.y = (unsigned)regs[i][2] | ((unsigned)regs[i][3] << 16);
      p.z = (unsigned)regs[i][4] | ((unsigned)regs[i][5] << 16);
      p.w = (unsigned)regs[i][6] | ((unsigned)regs[i][7] << 16);
      *(U4*)&Bs[buf][n][ko] = p;                 // one ds_store_b128
    }
  };
  auto bfrag = [&](int buf, int nt) -> v8u {
    v8u bu;
    int bn = nt * 16 + lr;
#pragma unroll
    for (int j = 0; j < 8; j++)
      bu[j] = *(const unsigned int*)&Bs[buf][bn][(j >> 2) * 16 + hf * 8 + 2 * (j & 3)];
    return bu;
  };

  v8f acc[NT];
#pragma unroll
  for (int nt = 0; nt < NT; nt++)
#pragma unroll
    for (int j = 0; j < 8; j++) acc[nt][j] = 0.f;

  // ---- prologue: stage tile 0 into buffer 0
  u16 rB[NOCT][8];
  asyncA(0, 0);
  loadB(0, rB);
  storeB(0, rB);
  asm volatile("s_wait_asynccnt 0x0" ::: "memory");
  __syncthreads();

  const int nsteps = K >> 5;
  for (int s = 0; s < nsteps; s++) {
    const int kk = s << 5;
    const int cur = s & 1, nxt = cur ^ 1;
    const bool hasNext = (s + 1) < nsteps;

    if (hasNext) {
      asyncA(kk + 32, nxt);         // overlaps with WMMA chain below
      loadB(kk + 32, rB);           // global loads in flight during compute
      if (s + 2 < nsteps)           // keep L2 warm two tiles ahead
        __builtin_prefetch(Ap + (long)z * sA + (long)(m0 + (t >> 1)) * K + kk + 64, 0, 1);
    }

    // ---- A fragment (16x32 bf16, ISA lane/VGPR packing) -> 2 x ds_load_b128
    v8u au;
    int ar = wv * 16 + lr;
#pragma unroll
    for (int j = 0; j < 4; j++) {
      int k0 = hf * 8 + 2 * j;
      au[j]     = *(const unsigned int*)&As[cur][ar][k0];
      au[4 + j] = *(const unsigned int*)&As[cur][ar][16 + k0];
    }
    v16bf a = __builtin_bit_cast(v16bf, au);

    // ---- WMMA chain, B fragment for nt+1 loads while nt issues
    v8u b0 = bfrag(cur, 0);
#pragma unroll
    for (int nt = 0; nt < NT; nt++) {
      v8u b1 = b0;
      if (nt + 1 < NT) b1 = bfrag(cur, nt + 1);
      acc[nt] = __builtin_amdgcn_wmma_f32_16x16x32_bf16(
          false, a, false, __builtin_bit_cast(v16bf, b0), (short)0, acc[nt],
          false, false);
      b0 = b1;
    }

    if (hasNext) {
      storeB(nxt, rB);
      asm volatile("s_wait_asynccnt 0x0" ::: "memory");
      __syncthreads();              // next buffer fully staged & visible
    }
  }

  // ---- epilogue: C layout = VGPR r -> row (hf*8 + r), lane lr -> col
#pragma unroll
  for (int nt = 0; nt < NT; nt++) {
    int col = n0 + nt * 16 + lr;
#pragma unroll
    for (int r = 0; r < 8; r++) {
      int row = m0 + wv * 16 + hf * 8 + r;
      long off = (long)z * sC + (long)row * N + col;
      float v = alpha * acc[nt][r];
      if (Rp) v += Rp[off];
      Cp[off] = v;
    }
  }
}

// ---------------------------------------------------------------------------
// LayerNorm over the feature axis (axis=2). One thread per (c, w) column:
// consecutive threads -> consecutive w -> coalesced vmem.
// ---------------------------------------------------------------------------
__global__ void ln_kernel(const float* __restrict__ x, const float* __restrict__ w,
                          const float* __restrict__ b, u16* __restrict__ out) {
  int c = blockIdx.y;
  int wc = blockIdx.x * blockDim.x + threadIdx.x;
  const float* xc = x + (long)c * FEATD * WDIM + wc;
  float s = 0.f, s2 = 0.f;
  for (int f = 0; f < FEATD; f++) {
    float v = xc[(long)f * WDIM];
    s += v; s2 += v * v;
  }
  float mu = s * (1.f / FEATD);
  float var = s2 * (1.f / FEATD) - mu * mu;
  float rstd = rsqrtf(var + 1e-5f);
  const float* wcp = w + (long)c * FEATD;
  const float* bcp = b + (long)c * FEATD;
  u16* oc = out + (long)c * FEATD * WDIM + wc;
  for (int f = 0; f < FEATD; f++) {
    float v = (xc[(long)f * WDIM] - mu) * rstd * wcp[f] + bcp[f];
    oc[(long)f * WDIM] = f2bf(v);
  }
}

// ---------------------------------------------------------------------------
// Channel mix: y[d,m,w] = sum_c dw[d,c] * in[c,m,w]
// mode 0: f32 out; mode 1: f32 out + res; mode 2: relu(x)^2 -> bf16 out
// ---------------------------------------------------------------------------
__global__ void dmix_kernel(const float* __restrict__ in, const float* __restrict__ dwm,
                            void* __restrict__ out, const float* __restrict__ res,
                            long MW, int mode) {
  long i = (long)blockIdx.x * blockDim.x + threadIdx.x;
  if (i >= MW) return;
  float v[CDIM];
#pragma unroll
  for (int c = 0; c < CDIM; c++) v[c] = in[(long)c * MW + i];
#pragma unroll
  for (int d = 0; d < CDIM; d++) {
    float acc = 0.f;
#pragma unroll
    for (int c = 0; c < CDIM; c++) acc += dwm[d * CDIM + c] * v[c];
    long off = (long)d * MW + i;
    if (mode == 2) {
      float r = acc > 0.f ? acc : 0.f;
      ((u16*)out)[off] = f2bf(r * r);
    } else {
      float o = acc;
      if (mode == 1) o += res[off];
      ((float*)out)[off] = o;
    }
  }
}

// ---------------------------------------------------------------------------
// Head split (+ optional RoPE): in [C][512][W] f32 -> out [C][NH][W][64] bf16
// ---------------------------------------------------------------------------
__global__ void rope_split_kernel(const float* __restrict__ in, u16* __restrict__ out,
                                  int do_rope) {
  long i = (long)blockIdx.x * blockDim.x + threadIdx.x;
  int pair = (int)(i & 31); long t = i >> 5;
  int w = (int)(t & (WDIM - 1)); t >>= 10;
  int h = (int)(t & (NHD - 1)); int c = (int)(t >> 3);
  int f0 = h * HD + 2 * pair;
  float x1 = in[((long)c * FEATD + f0) * WDIM + w];
  float x2 = in[((long)c * FEATD + f0 + 1) * WDIM + w];
  float o1, o2;
  if (do_rope) {
    float inv = __powf(10000.f, -(float)(2 * pair) / (float)HD);
    float ang = (float)w * inv;
    float sn, cs;
    __sincosf(ang, &sn, &cs);
    o1 = x1 * cs - x2 * sn;
    o2 = x2 * cs + x1 * sn;
  } else {
    o1 = x1; o2 = x2;
  }
  long ob = (((long)(c * NHD + h) * WDIM) + w) * HD + 2 * pair;
  out[ob] = f2bf(o1);
  out[ob + 1] = f2bf(o2);
}

// ---------------------------------------------------------------------------
// Row softmax: scores f32 [rows][W] -> probs bf16. One block per row.
// ---------------------------------------------------------------------------
__global__ void softmax_kernel(const float* __restrict__ sc, u16* __restrict__ pr, int Wn) {
  __shared__ float red[256];
  long row = blockIdx.x;
  const float* r = sc + row * (long)Wn;
  u16* o = pr + row * (long)Wn;
  int t = threadIdx.x;
  float m = -3.4e38f;
  for (int i = t; i < Wn; i += 256) m = fmaxf(m, r[i]);
  red[t] = m; __syncthreads();
  for (int s = 128; s > 0; s >>= 1) {
    if (t < s) red[t] = fmaxf(red[t], red[t + s]);
    __syncthreads();
  }
  m = red[0]; __syncthreads();
  float sum = 0.f;
  for (int i = t; i < Wn; i += 256) sum += __expf(r[i] - m);
  red[t] = sum; __syncthreads();
  for (int s = 128; s > 0; s >>= 1) {
    if (t < s) red[t] += red[t + s];
    __syncthreads();
  }
  float inv = 1.f / red[0];
  for (int i = t; i < Wn; i += 256) o[i] = f2bf(__expf(r[i] - m) * inv);
}

// ---------------------------------------------------------------------------
// Merge heads: av f32 [NH][W][64] (one channel) -> out bf16 [512][W]
// ---------------------------------------------------------------------------
__global__ void merge_kernel(const float* __restrict__ av, u16* __restrict__ out) {
  long i = (long)blockIdx.x * blockDim.x + threadIdx.x;
  int dh = (int)(i & (HD - 1)); long t = i >> 6;
  int w = (int)(t & (WDIM - 1)); int h = (int)(t >> 10);
  out[((long)(h * HD + dh)) * WDIM + w] = f2bf(av[i]);
}

__global__ void cast_bf16_kernel(const float* __restrict__ in, u16* __restrict__ out, long n) {
  long i = (long)blockIdx.x * blockDim.x + threadIdx.x;
  if (i < n) out[i] = f2bf(in[i]);
}

// ---------------------------------------------------------------------------
extern "C" void kernel_launch(void* const* d_in, const int* in_sizes, int n_in,
                              void* d_out, int out_size, void* d_ws, size_t ws_size,
                              hipStream_t stream) {
  (void)in_sizes; (void)n_in; (void)out_size; (void)ws_size;
  const float* x    = (const float*)d_in[0];
  const float* skip = (const float*)d_in[1];
  const float* n1w  = (const float*)d_in[2];
  const float* n1b  = (const float*)d_in[3];
  const float* n2w  = (const float*)d_in[4];
  const float* n2b  = (const float*)d_in[5];
  const float* n3w  = (const float*)d_in[6];
  const float* n3b  = (const float*)d_in[7];
  const float* aw[16];
  for (int i = 0; i < 16; i++) aw[i] = (const float*)d_in[8 + i];
  const float* c1pw = (const float*)d_in[24];
  const float* c1dw = (const float*)d_in[25];
  const float* c2pw = (const float*)d_in[26];
  float* out = (float*)d_out;

  // ---- workspace layout
  char* wsb = (char*)d_ws;
  size_t off = 0;
  auto alloc = [&](size_t bytes) -> char* {
    char* p = wsb + off;
    off = (off + bytes + 255) & ~(size_t)255;
    return p;
  };
  const long FW   = (long)FEATD * WDIM;     // 524288
  const long CFW  = (long)CDIM * FW;        // 4194304
  const long PWSZ = (long)CDIM * FEATD * FEATD;   // 2097152 per projection
  u16*   zb    = (u16*)  alloc(CFW * 2);
  u16*   skipb = (u16*)  alloc(CFW * 2);
  float* t0    = (float*)alloc(CFW * 4);
  float* qm    = (float*)alloc(CFW * 4);
  float* km    = (float*)alloc(CFW * 4);
  float* vm    = (float*)alloc(CFW * 4);
  u16*   qr    = (u16*)  alloc(CFW * 2);
  u16*   kr    = (u16*)  alloc(CFW * 2);
  u16*   vr    = (u16*)  alloc(CFW * 2);
  float* sc    = (float*)alloc((long)NHD * WDIM * WDIM * 4);
  u16*   pr    = (u16*)  alloc((long)NHD * WDIM * WDIM * 2);
  float* av    = (float*)alloc((long)NHD * WDIM * HD * 4);
  u16*   az    = (u16*)  alloc(CFW * 2);
  float* hb    = (float*)alloc(CFW * 4);
  float* t1    = (float*)alloc((long)CDIM * GEXP * WDIM * 4);
  u16*   ub    = (u16*)  alloc((long)CDIM * GEXP * WDIM * 2);
  // bf16 copies of all GEMM weights (so GEMM A staging is a pure async copy)
  u16* wpb[8];
  for (int j = 0; j < 8; j++) wpb[j] = (u16*)alloc(PWSZ * 2);
  u16* c1b = (u16*)alloc((long)CDIM * GEXP * FEATD * 2);
  u16* c2b = (u16*)alloc((long)CDIM * FEATD * GEXP * 2);

  dim3 b256(256);
  auto cast = [&](const float* src, u16* dst, long n) {
    cast_bf16_kernel<<<dim3((unsigned)((n + 255) / 256)), b256, 0, stream>>>(src, dst, n);
  };

  cast(skip, skipb, CFW);
  for (int j = 0; j < 8; j++) cast(aw[2 * j], wpb[j], PWSZ);   // a1q,a1k,a1v,a1o,a2q,...
  cast(c1pw, c1b, (long)CDIM * GEXP * FEATD);
  cast(c2pw, c2b, (long)CDIM * FEATD * GEXP);

  auto ln = [&](const float* src, const float* w, const float* bia, u16* dst) {
    ln_kernel<<<dim3(WDIM / 64, CDIM), dim3(64), 0, stream>>>(src, w, bia, dst);
  };
  auto gemm_w = [&](const u16* A, const u16* B, float* Cq, const float* R,
                    int M, int K, float alpha) {
    gemm_wmma<false, 8><<<dim3(WDIM / 128, M / 128, CDIM), b256, 0, stream>>>(
        A, B, Cq, R, M, WDIM, K, (long)M * K, (long)K * WDIM, (long)M * WDIM, alpha);
  };
  auto dmix = [&](const float* in, const float* dw, void* o, const float* r,
                  long MW, int mode) {
    dmix_kernel<<<dim3((unsigned)((MW + 255) / 256)), b256, 0, stream>>>(in, dw, o, r, MW, mode);
  };

  auto attn = [&](const u16* qsrc, const u16* msrc, u16* const* pw4,
                  const float* const* dw4, const float* res, float* hout) {
    // pw4: q,k,v,o bf16 projection weights; dw4: q,k,v,o channel-mix weights
    gemm_w(pw4[0], qsrc, t0, nullptr, FEATD, FEATD, 1.f);
    dmix(t0, dw4[0], qm, nullptr, FW, 0);
    gemm_w(pw4[1], msrc, t0, nullptr, FEATD, FEATD, 1.f);
    dmix(t0, dw4[1], km, nullptr, FW, 0);
    gemm_w(pw4[2], msrc, t0, nullptr, FEATD, FEATD, 1.f);
    dmix(t0, dw4[2], vm, nullptr, FW, 0);
    long tot = (long)CDIM * NHD * WDIM * (HD / 2);
    rope_split_kernel<<<dim3((unsigned)(tot / 256)), b256, 0, stream>>>(qm, qr, 1);
    rope_split_kernel<<<dim3((unsigned)(tot / 256)), b256, 0, stream>>>(km, kr, 1);
    rope_split_kernel<<<dim3((unsigned)(tot / 256)), b256, 0, stream>>>(vm, vr, 0);
    for (int c = 0; c < CDIM; c++) {
      const u16* qc = qr + (long)c * NHD * WDIM * HD;
      const u16* kc = kr + (long)c * NHD * WDIM * HD;
      const u16* vc = vr + (long)c * NHD * WDIM * HD;
      // scores = (q @ k^T) / sqrt(64)   (B given as N x K row-major)
      gemm_wmma<true, 8><<<dim3(WDIM / 128, WDIM / 128, NHD), b256, 0, stream>>>(
          qc, kc, sc, nullptr, WDIM, WDIM, HD,
          (long)WDIM * HD, (long)WDIM * HD, (long)WDIM * WDIM, 0.125f);
      softmax_kernel<<<dim3(NHD * WDIM), b256, 0, stream>>>(sc, pr, WDIM);
      // av = probs @ v
      gemm_wmma<false, 4><<<dim3(HD / 64, WDIM / 128, NHD), b256, 0, stream>>>(
          pr, vc, av, nullptr, WDIM, HD, WDIM,
          (long)WDIM * WDIM, (long)WDIM * HD, (long)WDIM * HD, 1.f);
      merge_kernel<<<dim3((NHD * WDIM * HD) / 256), b256, 0, stream>>>(av, az + (long)c * FW);
    }
    gemm_w(pw4[3], az, t0, nullptr, FEATD, FEATD, 1.f);
    dmix(t0, dw4[3], hout, res, FW, 1);
  };

  u16* pw1[4] = {wpb[0], wpb[1], wpb[2], wpb[3]};
  u16* pw2[4] = {wpb[4], wpb[5], wpb[6], wpb[7]};
  const float* dw1[4] = {aw[1], aw[3], aw[5], aw[7]};
  const float* dw2[4] = {aw[9], aw[11], aw[13], aw[15]};

  // ---- forward
  ln(x, n1w, n1b, zb);
  attn(zb, zb, pw1, dw1, x, hb);        // h = x + attn1(z, z)
  ln(hb, n2w, n2b, zb);
  attn(zb, skipb, pw2, dw2, hb, hb);    // h = h + attn2(z, skip)
  ln(hb, n3w, n3b, zb);
  // FFN: u = relu(dw-mix(c1pw @ z))^2 ; out = h + c2pw @ u
  gemm_wmma<false, 8><<<dim3(WDIM / 128, GEXP / 128, CDIM), b256, 0, stream>>>(
      c1b, zb, t1, nullptr, GEXP, WDIM, FEATD,
      (long)GEXP * FEATD, (long)FEATD * WDIM, (long)GEXP * WDIM, 1.f);
  dmix(t1, c1dw, ub, nullptr, (long)GEXP * WDIM, 2);
  gemm_wmma<false, 8><<<dim3(WDIM / 128, FEATD / 128, CDIM), b256, 0, stream>>>(
      c2b, ub, out, hb, FEATD, WDIM, GEXP,
      (long)FEATD * GEXP, (long)GEXP * WDIM, (long)FEATD * WDIM, 1.f);
}
